// LowFreqNoiseByHighFreqEmbeddings_2680059592965
// MI455X (gfx1250) — matrix operand order repaired and checked by reading, not compiled, not run on hardware
//
#include <hip/hip_runtime.h>
#include <math.h>

// ---------------------------------------------------------------------------
// LowFreqNoiseByHighFreqEmbeddings for MI455X (gfx1250, wave32, WMMA)
//
// Sizes (from reference): BS=4, SEQ=2048, VOCAB=32000, L=H=4096, D=1024.
// Core matrix op: cos = norm(high_e) @ norm(low_e)^T  (4096x4096x1024),
// fused with per-column (over h) masked running argmax so the full cos
// matrix is never materialized. Runs on v_wmma_f32_16x16x32_bf16 with
// 4-way h register blocking (4 WMMA per B fragment) and the block's
// 16-column B panel staged once in LDS (ds_load_b128 in the hot loop).
// ---------------------------------------------------------------------------

#define D_MODEL 1024
#define VOCAB   32000
#define BS      4
#define SEQ     2048
#define LOWN    4096
#define HIGHN   4096

#define BROW    1032   // LDS row pitch in ushorts (1024 + 8 pad -> conflict-free)

typedef __attribute__((ext_vector_type(8)))  float  v8f;
typedef __attribute__((ext_vector_type(16))) __bf16 v16bf;

union BV {            // 16 bf16 operand = 32 bytes = two 16B loads
    v16bf v;
    uint4 u[2];
};

__device__ __forceinline__ unsigned short f2bf(float f) {
    unsigned int u = __float_as_uint(f);
    unsigned int r = u + 0x7FFFu + ((u >> 16) & 1u);   // round-to-nearest-even
    return (unsigned short)(r >> 16);
}

// ---------------------------------------------------------------------------
// Kernel 1: gather lut rows for low/high ids, L2-normalize, store bf16.
// grid = 8192 blocks (4096 low rows then 4096 high rows), 256 threads.
// ---------------------------------------------------------------------------
__global__ __launch_bounds__(256)
void k_gather_normalize(const int* __restrict__ low_ids,
                        const int* __restrict__ high_ids,
                        const float* __restrict__ lut,
                        unsigned short* __restrict__ lowN,
                        unsigned short* __restrict__ highN) {
    int r = blockIdx.x;
    int id;
    unsigned short* dst;
    if (r < LOWN) { id = low_ids[r];          dst = lowN  + (size_t)r * D_MODEL; }
    else          { id = high_ids[r - LOWN];  dst = highN + (size_t)(r - LOWN) * D_MODEL; }

    const float* src = lut + (size_t)id * D_MODEL;
    int d = threadIdx.x * 4;
    float4 v = *(const float4*)(src + d);
    float ss = v.x * v.x + v.y * v.y + v.z * v.z + v.w * v.w;

    // wave32 reduction then cross-wave via LDS
    for (int off = 16; off > 0; off >>= 1) ss += __shfl_xor(ss, off);
    __shared__ float sred[8];
    int wave = threadIdx.x >> 5, lane = threadIdx.x & 31;
    if (lane == 0) sred[wave] = ss;
    __syncthreads();
    float tot = 0.f;
#pragma unroll
    for (int w = 0; w < 8; ++w) tot += sred[w];

    float rinv = (tot > 0.f) ? (1.0f / sqrtf(tot)) : 0.f;
    ushort4 o;
    o.x = f2bf(v.x * rinv); o.y = f2bf(v.y * rinv);
    o.z = f2bf(v.z * rinv); o.w = f2bf(v.w * rinv);
    *(ushort4*)(dst + d) = o;
}

// ---------------------------------------------------------------------------
// Kernel 2: pack [BS][4096] byte masks into [BS][128] uint32 bit words.
// (numpy bool_ is 1 byte; we read raw bytes.)
// ---------------------------------------------------------------------------
__global__ void k_pack_masks(const unsigned char* __restrict__ lowm,
                             const unsigned char* __restrict__ highm,
                             unsigned* __restrict__ lbits,
                             unsigned* __restrict__ hbits) {
    int w = blockIdx.x * blockDim.x + threadIdx.x;   // 0 .. 1023
    if (w >= 2 * BS * 128) return;
    int which = w >> 9;            // 0 = low, 1 = high (512 words each)
    int idx   = w & 511;
    const unsigned char* src = which ? highm : lowm;
    unsigned bits = 0;
#pragma unroll
    for (int i = 0; i < 32; ++i)
        bits |= (unsigned)(src[idx * 32 + i] != 0) << i;
    (which ? hbits : lbits)[idx] = bits;
}

// ---------------------------------------------------------------------------
// Kernel 3: fused GEMM + masked running argmax.
// Block = one 16-wide l-panel (B panel staged in LDS, 33KB, row pitch BROW),
// 8 waves; wave w covers h in [w*512, (w+1)*512) as 8 groups of 4 WMMA tiles
// (4-way h register blocking: one ds-loaded B fragment feeds 4 WMMAs).
// C layout (32-bit 16x16): VGPR r, lane c -> M = r + 8*(lane>=16), N = lane&15.
// A (16x32 bf16): lane m=lane&15 holds row M=m; lanes<16: K{0..7,16..23},
// lanes>=16: K{8..15,24..31}. B (32x16 bf16): lane holds col N=lane&15,
// same K split as A (lanes<16: K{0..7,16..23}, lanes>=16: K{8..15,24..31}).
// ---------------------------------------------------------------------------
__global__ __launch_bounds__(256)
void k_argmax_gemm(const unsigned short* __restrict__ lowN,
                   const unsigned short* __restrict__ highN,
                   const unsigned* __restrict__ hbits,
                   const unsigned* __restrict__ lbits,
                   float* __restrict__ bestv_out,
                   int*   __restrict__ besti_out) {
    const int l0   = blockIdx.x * 16;
    const int wave = threadIdx.x >> 5;
    const int lane = threadIdx.x & 31;
    const int half = lane >> 4;      // 0 or 1
    const int m    = lane & 15;

    // ---- stage the 16-row B panel (lowN rows l0..l0+15) into LDS ----
    __shared__ unsigned short smem[16 * BROW];
    {
        int r  = threadIdx.x >> 4;         // 16 threads per row
        int cg = threadIdx.x & 15;         // each covers 64 ushorts = 8 uint4
        const unsigned short* g = lowN + (size_t)(l0 + r) * D_MODEL + cg * 64;
        unsigned short* s = smem + r * BROW + cg * 64;
#pragma unroll
        for (int i = 0; i < 8; ++i)
            *(uint4*)(s + i * 8) = *(const uint4*)(g + i * 8);
    }
    __syncthreads();

    const unsigned short* brow = smem + m * BROW + half * 8;

    float bv[BS]; int bi[BS];
#pragma unroll
    for (int b = 0; b < BS; ++b) { bv[b] = -INFINITY; bi[b] = 0; }

    for (int t = 0; t < 8; ++t) {
        const int hbase = wave * 512 + t * 64;
        const unsigned short* arow = highN + (size_t)(hbase + m) * D_MODEL + half * 8;

        v8f c[4] = {{}, {}, {}, {}};
#pragma unroll 2
        for (int k0 = 0; k0 < D_MODEL; k0 += 32) {
            BV b2;
            b2.u[0] = *(const uint4*)(brow + k0);
            b2.u[1] = *(const uint4*)(brow + k0 + 16);
#pragma unroll
            for (int j = 0; j < 4; ++j) {
                BV a;
                const unsigned short* ap = arow + j * (16 * D_MODEL) + k0;
                a.u[0] = *(const uint4*)(ap);
                a.u[1] = *(const uint4*)(ap + 16);
                c[j] = __builtin_amdgcn_wmma_f32_16x16x32_bf16(
                           false, a.v, false, b2.v, (short)0, c[j], false, false);
            }
        }

        // masked running argmax per batch (first-max tie rule via strict >)
#pragma unroll
        for (int j = 0; j < 4; ++j) {
            const int h0 = hbase + j * 16;
#pragma unroll
            for (int b = 0; b < BS; ++b) {
                unsigned word = hbits[b * 128 + (h0 >> 5)];
                unsigned base = (unsigned)((h0 & 31) + half * 8);
#pragma unroll
                for (int r = 0; r < 8; ++r) {
                    float v = c[j][r];
                    int   h = h0 + half * 8 + r;
                    bool masked = (word >> (base + r)) & 1u;
                    if (!masked && v > bv[b]) { bv[b] = v; bi[b] = h; }
                }
            }
        }
    }

    // combine the two lanes holding the same column (h sets disjoint, ordered)
#pragma unroll
    for (int b = 0; b < BS; ++b) {
        float ov = __shfl_xor(bv[b], 16);
        int   oi = __shfl_xor(bi[b], 16);
        if (ov > bv[b] || (ov == bv[b] && oi < bi[b])) { bv[b] = ov; bi[b] = oi; }
    }

    __shared__ float sV[8][16][BS];
    __shared__ int   sI[8][16][BS];
    if (half == 0) {
#pragma unroll
        for (int b = 0; b < BS; ++b) { sV[wave][m][b] = bv[b]; sI[wave][m][b] = bi[b]; }
    }
    __syncthreads();

    // cross-wave reduction: 64 threads, one per (l, batch)
    int tid = threadIdx.x;
    if (tid < 16 * BS) {
        int b  = tid & (BS - 1);
        int ml = tid >> 2;
        float best = sV[0][ml][b];
        int   bidx = sI[0][ml][b];
#pragma unroll
        for (int w = 1; w < 8; ++w) {
            float v = sV[w][ml][b];
            int   i = sI[w][ml][b];
            if (v > best || (v == best && i < bidx)) { best = v; bidx = i; }
        }
        int l = l0 + ml;
        unsigned lw = lbits[b * 128 + (l >> 5)];
        if ((lw >> (l & 31)) & 1u) best = -INFINITY;   // low-masked column
        bestv_out[b * LOWN + l] = best;
        besti_out[b * LOWN + l] = bidx;
    }
}

// ---------------------------------------------------------------------------
// Kernel 4: softmax denominator Z[b][h] = sum_l exp(v) over routed columns.
// (logits are cosines in ~[-1,1]; max-subtraction is unnecessary for range.)
// ---------------------------------------------------------------------------
__global__ void k_z_accum(const float* __restrict__ bestv,
                          const int*   __restrict__ besti,
                          float* __restrict__ Z) {
    int tid = blockIdx.x * blockDim.x + threadIdx.x;
    if (tid >= BS * LOWN) return;
    float v = bestv[tid];
    if (v > -1e30f) {
        int b = tid >> 12;
        atomicAdd(&Z[b * HIGHN + besti[tid]], expf(v));
    }
}

// ---------------------------------------------------------------------------
// Kernel 5: scatter  noise[b, h] += (exp(v)/Z) * lut[low_ids[l]].
// One block per (b,l); 256 threads x 4 floats; atomic f32 adds (L2 atomics).
// ---------------------------------------------------------------------------
__global__ __launch_bounds__(256)
void k_noise_scatter(const float* __restrict__ bestv,
                     const int*   __restrict__ besti,
                     const float* __restrict__ Z,
                     const int*   __restrict__ low_ids,
                     const float* __restrict__ lut,
                     float* __restrict__ noise) {
    int p = blockIdx.x;              // 0 .. BS*LOWN-1
    float v = bestv[p];
    if (v <= -1e30f) return;
    int b = p >> 12, l = p & (LOWN - 1);
    int h = besti[p];
    float w = expf(v) / Z[b * HIGHN + h];
    const float* src = lut + (size_t)low_ids[l] * D_MODEL;
    float* dst = noise + ((size_t)b * (HIGHN + 1) + h) * D_MODEL;
    int d = threadIdx.x * 4;
#pragma unroll
    for (int i = 0; i < 4; ++i)
        atomicAdd(&dst[d + i], w * src[d + i]);
}

// ---------------------------------------------------------------------------
// Kernel 6: out[b,s,:] = (lut[x] + noise[b, rhf[b,x]]) * sqrt(D) ; sqrt=32.
// ---------------------------------------------------------------------------
__global__ __launch_bounds__(256)
void k_final_gather(const int* __restrict__ x,
                    const int* __restrict__ rhf,
                    const float* __restrict__ lut,
                    const float* __restrict__ noise,
                    float* __restrict__ out) {
    int p = blockIdx.x;              // 0 .. BS*SEQ-1
    int b = p >> 11;
    int tok = x[p];
    int idx = rhf[b * VOCAB + tok];
    if ((unsigned)idx > (unsigned)HIGHN) idx = HIGHN;   // <0 or OOB -> zero row
    int d = threadIdx.x * 4;
    float4 lv = *(const float4*)(lut + (size_t)tok * D_MODEL + d);
    float4 nv = *(const float4*)(noise + ((size_t)b * (HIGHN + 1) + idx) * D_MODEL + d);
    float4 o;
    o.x = (lv.x + nv.x) * 32.0f;
    o.y = (lv.y + nv.y) * 32.0f;
    o.z = (lv.z + nv.z) * 32.0f;
    o.w = (lv.w + nv.w) * 32.0f;
    *(float4*)(out + (size_t)p * D_MODEL + d) = o;
}

// ---------------------------------------------------------------------------
extern "C" void kernel_launch(void* const* d_in, const int* in_sizes, int n_in,
                              void* d_out, int out_size, void* d_ws, size_t ws_size,
                              hipStream_t stream) {
    const int*           x        = (const int*)d_in[0];            // [BS, SEQ]
    const int*           rhf      = (const int*)d_in[1];            // [BS, VOCAB]
    const int*           low_ids  = (const int*)d_in[2];            // [LOWN]
    const int*           high_ids = (const int*)d_in[3];            // [HIGHN]
    const unsigned char* lowm     = (const unsigned char*)d_in[4];  // [BS, LOWN] bool
    const unsigned char* highm    = (const unsigned char*)d_in[5];  // [BS, HIGHN] bool
    const float*         lut      = (const float*)d_in[6];          // [VOCAB, D]
    float*               out      = (float*)d_out;                  // [BS, SEQ, D]

    // workspace layout
    char* ws = (char*)d_ws;
    size_t off = 0;
    unsigned short* lowN  = (unsigned short*)(ws + off); off += (size_t)LOWN  * D_MODEL * 2;
    unsigned short* highN = (unsigned short*)(ws + off); off += (size_t)HIGHN * D_MODEL * 2;
    unsigned* hbits = (unsigned*)(ws + off); off += BS * 128 * 4;
    unsigned* lbits = (unsigned*)(ws + off); off += BS * 128 * 4;
    float* bestv = (float*)(ws + off); off += (size_t)BS * LOWN * 4;
    int*   besti = (int*)  (ws + off); off += (size_t)BS * LOWN * 4;
    float* Z     = (float*)(ws + off); off += (size_t)BS * HIGHN * 4;
    float* noise = (float*)(ws + off); off += (size_t)BS * (HIGHN + 1) * D_MODEL * 4;

    hipMemsetAsync(Z, 0, (size_t)BS * HIGHN * 4, stream);
    hipMemsetAsync(noise, 0, (size_t)BS * (HIGHN + 1) * D_MODEL * 4, stream);

    k_gather_normalize<<<LOWN + HIGHN, 256, 0, stream>>>(low_ids, high_ids, lut, lowN, highN);
    k_pack_masks<<<4, 256, 0, stream>>>(lowm, highm, lbits, hbits);
    k_argmax_gemm<<<LOWN / 16, 256, 0, stream>>>(lowN, highN, hbits, lbits, bestv, besti);
    k_z_accum<<<(BS * LOWN + 255) / 256, 256, 0, stream>>>(bestv, besti, Z);
    k_noise_scatter<<<BS * LOWN, 256, 0, stream>>>(bestv, besti, Z, low_ids, lut, noise);
    k_final_gather<<<BS * SEQ, 256, 0, stream>>>(x, rhf, lut, noise, out);
}
